// ModelFC_49134425866355
// MI455X (gfx1250) — matrix operand order repaired and checked by reading, not compiled, not run on hardware
//
#include <hip/hip_runtime.h>
#include <hip/hip_bf16.h>
#include <math.h>

typedef __attribute__((ext_vector_type(16))) _Float16 v16h;
typedef __attribute__((ext_vector_type(8)))  _Float16 v8h;
typedef __attribute__((ext_vector_type(8)))  float    v8f;

// Problem constants (static per reference)
#define A_DIM   64
#define EPC     8
#define ETOT    512          // A_DIM * EPC
#define NROWS   32768        // A_DIM * ETOT
#define H       768
#define HIDIM   1536
#define DOUT    100
#define DPAD    112          // padded logits stride (7 tiles of 16)
// LDS h stride in halves: row byte offset 1544*2=3088 (16B aligned for b128),
// bank offset per row = (1544/2) % 64 = 4 -> 16 rows on distinct banks.
#define HSTRIDE 1544

#define NT1     96           // HIDIM/16 n-tiles in GEMM1
#define KC1     24           // H/32 k-chunks in GEMM1
#define NT2     7            // DPAD/16 n-tiles in GEMM2
#define KC2     48           // HIDIM/32 k-chunks in GEMM2

#define SCORES_SZ (A_DIM * A_DIM * EPC * EPC)   // 262144

// LDS layout (bytes)
#define LDSA_BYTES  (KC1 * 32 * 16 * 2)                   // 24576
#define LDSH_BYTES  (16 * HSTRIDE * 2)                    // 49408
#define SMEM_BYTES  (LDSA_BYTES + LDSH_BYTES + (256 + 32) * 4)

// Workspace layout (bytes)
#define BSW1_BYTES ((size_t)NT1 * KC1 * 32 * 16 * 2)      // 2,359,296
#define BSW2_BYTES ((size_t)NT2 * KC2 * 32 * 16 * 2)      //   344,064
#define LOGITS_BYTES ((size_t)NROWS * DPAD * 4)           // 14,680,064

// ---------------------------------------------------------------------------
// W1 [H][HIDIM] f32 -> f16 B-fragments: [ntile][kchunk][lane][16 halves]
// B 32x16 layout: lane L holds column n = L&15; K = (L>>4)*8 + 2*v + w + (v>=4?8:0).
// ---------------------------------------------------------------------------
__global__ void swizzle_w1_kernel(const float* __restrict__ W1,
                                  _Float16* __restrict__ Bsw1) {
    int e = blockIdx.x * blockDim.x + threadIdx.x;   // (ntile*KC1 + c)*32 + L
    if (e >= NT1 * KC1 * 32) return;
    int L = e & 31;
    int c = (e >> 5) % KC1;
    int ntile = e / (KC1 * 32);
    int n  = ntile * 16 + (L & 15);
    int hi = L >> 4;
    _Float16* dst = Bsw1 + (size_t)e * 16;
#pragma unroll
    for (int v = 0; v < 8; ++v) {
#pragma unroll
        for (int w = 0; w < 2; ++w) {
            int kk = c * 32 + hi * 8 + 2 * v + w + ((v >= 4) ? 8 : 0);
            dst[2 * v + w] = (_Float16)W1[(size_t)kk * HIDIM + n];
        }
    }
}

// W2 [HIDIM][DOUT] f32 -> f16 B-fragments, columns >= DOUT zero padded
__global__ void swizzle_w2_kernel(const float* __restrict__ W2,
                                  _Float16* __restrict__ Bsw2) {
    int e = blockIdx.x * blockDim.x + threadIdx.x;   // (ntile*KC2 + c)*32 + L
    if (e >= NT2 * KC2 * 32) return;
    int L = e & 31;
    int c = (e >> 5) % KC2;
    int ntile = e / (KC2 * 32);
    int n  = ntile * 16 + (L & 15);
    int hi = L >> 4;
    _Float16* dst = Bsw2 + (size_t)e * 16;
#pragma unroll
    for (int v = 0; v < 8; ++v) {
#pragma unroll
        for (int w = 0; w < 2; ++w) {
            int kk = c * 32 + hi * 8 + 2 * v + w + ((v >= 4) ? 8 : 0);
            float val = (n < DOUT) ? W2[(size_t)kk * DOUT + n] : 0.0f;
            dst[2 * v + w] = (_Float16)val;
        }
    }
}

// ---------------------------------------------------------------------------
// Fused per-16-row tile: logits = (LN(GELU(E@W1+b1)) @ W2 + b2)
// block = 256 threads (8 waves), grid = NROWS/16 = 2048
// ---------------------------------------------------------------------------
__global__ __launch_bounds__(256) void fused_mlp_kernel(
    const float*   __restrict__ E,
    const _Float16* __restrict__ Bsw1,
    const float*   __restrict__ b1,
    const float*   __restrict__ ln_g,
    const float*   __restrict__ ln_b,
    const _Float16* __restrict__ Bsw2,
    const float*   __restrict__ b2,
    float*         __restrict__ logits) {
    extern __shared__ char smem[];
    _Float16* ldsA = (_Float16*)smem;                          // A fragments
    _Float16* ldsH = (_Float16*)(smem + LDSA_BYTES);           // h tile (f16)
    float*    red  = (float*)(smem + LDSA_BYTES + LDSH_BYTES); // 8 waves*16 rows*{sum,sq}
    float*    stats = red + 256;                               // mu[16], rstd[16]

    const int tid  = threadIdx.x;
    const int wave = tid >> 5;
    const int lane = tid & 31;
    const int hi   = lane >> 4;
    const int ln16 = lane & 15;
    const int rowBase = blockIdx.x * 16;

    // ---- Stage A tile (16 x 768 f32 -> f16) into WMMA A-fragment layout ----
    // A 16x32 layout: lane L holds row m = L&15; K = (L>>4)*8 + (2v + (v>=4?8:0)) + w
    for (int e = tid; e < KC1 * 32; e += 256) {
        int c   = e >> 5;
        int L   = e & 31;
        int r   = L & 15;
        int lhi = L >> 4;
        const float* src = E + (size_t)(rowBase + r) * H + c * 32 + lhi * 8;
        _Float16* dst = ldsA + (size_t)e * 16;
#pragma unroll
        for (int v = 0; v < 8; ++v) {
            int kk = 2 * v + ((v >= 4) ? 8 : 0);
            float2 xy = *(const float2*)(src + kk);
            dst[2 * v]     = (_Float16)xy.x;
            dst[2 * v + 1] = (_Float16)xy.y;
        }
    }
    __syncthreads();

    // ---- GEMM1 + bias + exact GELU; per-row sum/sumsq for LN ----
    float rsum[8], rsq[8];
#pragma unroll
    for (int r = 0; r < 8; ++r) { rsum[r] = 0.0f; rsq[r] = 0.0f; }

    for (int t = 0; t < 12; ++t) {
        int ntile = wave * 12 + t;
        v8f acc = {};
        const _Float16* bptr = Bsw1 + (size_t)ntile * KC1 * 32 * 16;
#pragma unroll 4
        for (int c = 0; c < KC1; ++c) {
            v16h a = *(const v16h*)(ldsA + ((size_t)c * 32 + lane) * 16);
            v16h b = *(const v16h*)(bptr + ((size_t)c * 32 + lane) * 16);
            acc = __builtin_amdgcn_wmma_f32_16x16x32_f16(
                false, a, false, b, (short)0, acc, false, false);
        }
        int n = ntile * 16 + ln16;
        float bias = b1[n];
#pragma unroll
        for (int r = 0; r < 8; ++r) {
            int m = r + hi * 8;                       // C layout: vgpr r, half hi
            float x   = acc[r] + bias;
            float gel = 0.5f * x * (1.0f + erff(x * 0.70710678118654752f));
            ldsH[(size_t)m * HSTRIDE + n] = (_Float16)gel;
            rsum[r] += gel;
            rsq[r]  += gel * gel;
        }
    }
    // reduce over the 16 lanes of each half-wave (columns of this wave)
#pragma unroll
    for (int r = 0; r < 8; ++r) {
        for (int s = 1; s < 16; s <<= 1) {
            rsum[r] += __shfl_xor(rsum[r], s, 32);
            rsq[r]  += __shfl_xor(rsq[r], s, 32);
        }
    }
    if (ln16 == 0) {
#pragma unroll
        for (int r = 0; r < 8; ++r) {
            int m = r + hi * 8;
            red[(wave * 16 + m) * 2 + 0] = rsum[r];
            red[(wave * 16 + m) * 2 + 1] = rsq[r];
        }
    }
    __syncthreads();
    if (tid < 16) {
        float s = 0.0f, q = 0.0f;
        for (int w = 0; w < 8; ++w) {
            s += red[(w * 16 + tid) * 2 + 0];
            q += red[(w * 16 + tid) * 2 + 1];
        }
        float mu  = s * (1.0f / HIDIM);
        float var = q * (1.0f / HIDIM) - mu * mu;
        stats[tid]      = mu;
        stats[16 + tid] = rsqrtf(var + 1e-12f);
    }
    __syncthreads();

    // ---- Apply LayerNorm affine in place on ldsH (vectorized, disjoint) ----
    for (int i = tid; i < 16 * (HIDIM / 2); i += 256) {
        int m  = i / (HIDIM / 2);
        int k2 = (i - m * (HIDIM / 2)) * 2;
        float mu = stats[m];
        float rs = stats[16 + m];
        _Float16* p = ldsH + (size_t)m * HSTRIDE + k2;
        float x0 = (float)p[0];
        float x1 = (float)p[1];
        p[0] = (_Float16)((x0 - mu) * rs * ln_g[k2]     + ln_b[k2]);
        p[1] = (_Float16)((x1 - mu) * rs * ln_g[k2 + 1] + ln_b[k2 + 1]);
    }
    __syncthreads();

    // ---- GEMM2: 7 waves each own one 16-col tile (covers DPAD=112 >= 100) ----
    if (wave < NT2) {
        v8f acc = {};
        const _Float16* bptr = Bsw2 + (size_t)wave * KC2 * 32 * 16;
        const _Float16* hrow = ldsH + (size_t)ln16 * HSTRIDE;
#pragma unroll 4
        for (int c = 0; c < KC2; ++c) {
            // A fragment: K pattern per half-wave = base+{0..7} and base+16+{0..7}
            // -> two contiguous 16B runs -> 2x ds_load_b128
            int base = c * 32 + hi * 8;
            v8h lo = *(const v8h*)(hrow + base);
            v8h hv = *(const v8h*)(hrow + base + 16);
            v16h a;
#pragma unroll
            for (int q = 0; q < 8; ++q) { a[q] = lo[q]; a[8 + q] = hv[q]; }
            v16h b = *(const v16h*)(bptr + ((size_t)c * 32 + lane) * 16);
            acc = __builtin_amdgcn_wmma_f32_16x16x32_f16(
                false, a, false, b, (short)0, acc, false, false);
        }
        int n = wave * 16 + ln16;
        if (n < DOUT) {
            float bias = b2[n];
#pragma unroll
            for (int r = 0; r < 8; ++r) {
                int m = r + hi * 8;
                logits[(size_t)(rowBase + m) * DPAD + n] = acc[r] + bias;
            }
        }
    }
}

// ---------------------------------------------------------------------------
// Per-entity own-row softmax max / argmax; writes VG_scores + indices; sel
// ---------------------------------------------------------------------------
__global__ void own_softmax_kernel(const float* __restrict__ logits,
                                   float* __restrict__ out,
                                   int* __restrict__ sel) {
    int e = blockIdx.x * blockDim.x + threadIdx.x;
    if (e >= ETOT) return;
    int step = e >> 3;                       // arange(Etot) // EP
    const float* row = logits + (size_t)(step * ETOT + e) * DPAD;
    float m = row[0];
    int am = 0;
    for (int d = 1; d < DOUT; ++d) {
        float v = row[d];
        if (v > m) { m = v; am = d; }        // strict '>' keeps first max (jnp.argmax)
    }
    float s = 0.0f;
    for (int d = 0; d < DOUT; ++d) s += expf(row[d] - m);
    out[SCORES_SZ + e]        = 1.0f / s;    // max(softmax) == exp(0)/sum
    out[SCORES_SZ + ETOT + e] = (float)am;
    if (e == ETOT - 1) *sel = am;
}

// ---------------------------------------------------------------------------
// scores[0][l][m][j][k] = logits[l*512 + m*8 + j][sel]  (broadcast over k)
// ---------------------------------------------------------------------------
__global__ void scores_bcast_kernel(const float* __restrict__ logits,
                                    const int* __restrict__ selp,
                                    float* __restrict__ out) {
    int idx = blockIdx.x * blockDim.x + threadIdx.x;
    if (idx >= SCORES_SZ) return;
    int j = (idx >> 3) & 7;
    int m = (idx >> 6) & 63;
    int l = idx >> 12;
    int sel = *selp;
    int row = l * ETOT + m * EPC + j;
    out[idx] = logits[(size_t)row * DPAD + sel];
}

// ---------------------------------------------------------------------------
extern "C" void kernel_launch(void* const* d_in, const int* in_sizes, int n_in,
                              void* d_out, int out_size, void* d_ws, size_t ws_size,
                              hipStream_t stream) {
    (void)in_sizes; (void)n_in; (void)out_size; (void)ws_size;
    const float* E    = (const float*)d_in[0];
    const float* W1   = (const float*)d_in[1];
    const float* b1   = (const float*)d_in[2];
    const float* ln_g = (const float*)d_in[3];
    const float* ln_b = (const float*)d_in[4];
    const float* W2   = (const float*)d_in[5];
    const float* b2   = (const float*)d_in[6];
    // d_in[7] = entity_count (static 8, baked into indexing)

    float* out = (float*)d_out;
    char*  ws  = (char*)d_ws;
    _Float16* Bsw1  = (_Float16*)ws;
    _Float16* Bsw2  = (_Float16*)(ws + BSW1_BYTES);
    float*   logits = (float*)(ws + BSW1_BYTES + BSW2_BYTES);
    int*     sel    = (int*)(ws + BSW1_BYTES + BSW2_BYTES + LOGITS_BYTES);

    swizzle_w1_kernel<<<(NT1 * KC1 * 32 + 255) / 256, 256, 0, stream>>>(W1, Bsw1);
    swizzle_w2_kernel<<<(NT2 * KC2 * 32 + 255) / 256, 256, 0, stream>>>(W2, Bsw2);

    fused_mlp_kernel<<<NROWS / 16, 256, SMEM_BYTES, stream>>>(
        E, Bsw1, b1, ln_g, ln_b, Bsw2, b2, logits);

    own_softmax_kernel<<<2, 256, 0, stream>>>(logits, out, sel);
    scores_bcast_kernel<<<(SCORES_SZ + 255) / 256, 256, 0, stream>>>(logits, sel, out);
}
